// CMambaBlockAdapted_40054865003109
// MI455X (gfx1250) — matrix operand order repaired and checked by reading, not compiled, not run on hardware
//
#include <hip/hip_runtime.h>
#include <hip/hip_bf16.h>
#include <stdint.h>

// ---- problem constants ----
#define BN    1024
#define LSEQ  64
#define DM    128
#define DFF   256
#define DS_   16
#define DTR_  16
#define EPSV  1e-5f

typedef __attribute__((ext_vector_type(16))) _Float16 v16h;
typedef __attribute__((ext_vector_type(8)))  float    v8f;

union V16H { v16h v; unsigned u[8]; _Float16 h[16]; };

// ---------------------------------------------------------------------------
// WMMA fragment loaders (CDNA5 V_WMMA_F32_16X16X32_F16 layouts, wave32)
// A (16x32, M x K): lane holds row m = lane%16; VGPR v, half p holds
//   K = k0 + 16*(lane/16) + 2v + p  -> dword-pair gathers.
// B (32x16, K x N): lane holds col n = lane%16; element e holds
//   K = k0 + 16*(lane/16) + e  -> 32 contiguous bytes per lane.
// C/D (16x16): VGPR r holds M = r + 8*(lane/16), N = lane%16.
// ---------------------------------------------------------------------------
template<int RL>
__device__ __forceinline__ v16h load_a_frag(const _Float16* p, int row0, int k0) {
  const int lane = threadIdx.x & 31;
  const int m  = row0 + (lane & 15);
  const int hs = lane >> 4;
  const unsigned* pu = (const unsigned*)(p + (size_t)m * RL);
  V16H r;
#pragma unroll
  for (int v = 0; v < 8; ++v) {
    int kb = 2 * v + 16 * hs + k0; // always even
    r.u[v] = pu[kb >> 1];
  }
  return r.v;
}

// W is row-major (N x K) f16 (torch-style weight); B[k][n] = W[n][k].
template<int RL>
__device__ __forceinline__ v16h load_b_frag(const _Float16* w, int n0, int k0) {
  const int lane = threadIdx.x & 31;
  const int n  = n0 + (lane & 15);
  const int kb = k0 + 16 * (lane >> 4);
  const uint4* p = (const uint4*)(w + (size_t)n * RL + kb);
  uint4 q0 = p[0], q1 = p[1];
  V16H r;
  r.u[0] = q0.x; r.u[1] = q0.y; r.u[2] = q0.z; r.u[3] = q0.w;
  r.u[4] = q1.x; r.u[5] = q1.y; r.u[6] = q1.z; r.u[7] = q1.w;
  return r.v;
}

__device__ __forceinline__ v8f wmma_f16(v16h a, v16h b, v8f c) {
  return __builtin_amdgcn_wmma_f32_16x16x32_f16(false, a, false, b, (short)0, c,
                                                false, false);
}

__device__ __forceinline__ float silu_f(float x) { return x / (1.f + __expf(-x)); }
__device__ __forceinline__ float softplus_f(float x) {
  return (x > 20.f) ? x : __logf(1.f + __expf(x));
}

// Async global -> LDS tile stage (ASYNCcnt path, 16B per lane per op).
// n16 = number of 16-byte chunks; all threads participate; caller must
// __syncthreads() after (we wait our own ASYNCcnt here).
__device__ __forceinline__ void async_stage_tile(const void* gsrc, void* lds_dst,
                                                 int n16) {
  const int tid = threadIdx.x;
  uint64_t g0 = (uint64_t)gsrc;
  unsigned l0 = (unsigned)(uintptr_t)lds_dst; // low 32 bits = LDS byte offset
  for (int i = tid; i < n16; i += 256) {
    unsigned lds = l0 + i * 16;
    uint64_t g = g0 + (uint64_t)i * 16;
    asm volatile("global_load_async_to_lds_b128 %0, %1, off"
                 :: "v"(lds), "v"(g) : "memory");
  }
  asm volatile("s_wait_asynccnt 0x0" ::: "memory");
}

// ---------------------------------------------------------------------------
// K0: f32 -> f16 weight conversion
// ---------------------------------------------------------------------------
__global__ void k_convert(const float* __restrict__ a, int n,
                          _Float16* __restrict__ o) {
  int i = blockIdx.x * 256 + threadIdx.x;
  if (i < n) o[i] = (_Float16)a[i];
}

// ---------------------------------------------------------------------------
// K1: RMSNorm + in_proj GEMM (64x512, K=128) + split + causal conv + SiLU
// one block per sequence (bn); 8 waves; wave w owns output cols [w*64, w*64+64)
// ---------------------------------------------------------------------------
__global__ __launch_bounds__(256) void k_fused_inproj(
    const float* __restrict__ x, const float* __restrict__ norm_w,
    const _Float16* __restrict__ Win, const float* __restrict__ in_b,
    const float* __restrict__ conv_w, const float* __restrict__ conv_b,
    _Float16* __restrict__ xc_g, _Float16* __restrict__ zs_g) {
  __shared__ _Float16 sh_h[LSEQ * DM];    // 16 KB normalized activations
  __shared__ _Float16 sh_xin[LSEQ * DFF]; // 32 KB pre-conv branch
  const int bn = blockIdx.x, tid = threadIdx.x;
  const int wv = tid >> 5, lane = tid & 31;
  const float* xr = x + (size_t)bn * LSEQ * DM;

  // RMSNorm: wave w handles rows w*8 .. w*8+7; lane covers 4 of 128 dims
  float4 nw = ((const float4*)norm_w)[lane];
#pragma unroll
  for (int i = 0; i < 8; ++i) {
    int r = wv * 8 + i;
    float4 v = ((const float4*)(xr + r * DM))[lane];
    float ss = v.x * v.x + v.y * v.y + v.z * v.z + v.w * v.w;
#pragma unroll
    for (int off = 16; off > 0; off >>= 1) ss += __shfl_xor(ss, off, 32);
    float sc = rsqrtf(ss * (1.f / DM) + EPSV);
    _Float16* hrow = sh_h + r * DM + lane * 4;
    hrow[0] = (_Float16)(v.x * sc * nw.x);
    hrow[1] = (_Float16)(v.y * sc * nw.y);
    hrow[2] = (_Float16)(v.z * sc * nw.z);
    hrow[3] = (_Float16)(v.w * sc * nw.w);
  }
  __syncthreads();

  // GEMM: 4 row-tiles x 4 col-tiles per wave, K = 128 (4 k-steps)
  const int n0 = wv * 64;
  v8f acc[4][4] = {};
  for (int kk = 0; kk < 4; ++kk) {
    const int k0 = kk * 32;
    v16h af[4];
#pragma unroll
    for (int rt = 0; rt < 4; ++rt) af[rt] = load_a_frag<DM>(sh_h, rt * 16, k0);
#pragma unroll
    for (int ct = 0; ct < 4; ++ct) {
      v16h b = load_b_frag<DM>(Win, n0 + ct * 16, k0);
#pragma unroll
      for (int rt = 0; rt < 4; ++rt) acc[rt][ct] = wmma_f16(af[rt], b, acc[rt][ct]);
    }
  }

  // epilogue: xin -> LDS, silu(z) -> HBM f16 (branch is wave-uniform)
  const int hs = lane >> 4, nn = lane & 15;
  _Float16* zrow = zs_g + (size_t)bn * LSEQ * DFF;
#pragma unroll
  for (int rt = 0; rt < 4; ++rt)
#pragma unroll
    for (int ct = 0; ct < 4; ++ct)
#pragma unroll
      for (int r = 0; r < 8; ++r) {
        int m = rt * 16 + r + 8 * hs;
        int n = n0 + ct * 16 + nn;
        float v = acc[rt][ct][r] + in_b[n];
        if (n < DFF) sh_xin[m * DFF + n] = (_Float16)v;
        else         zrow[m * DFF + (n - DFF)] = (_Float16)silu_f(v);
      }
  __syncthreads();

  // depthwise causal conv (taps=4) + SiLU; thread owns channel c = tid
  {
    const int c = tid;
    float w0 = conv_w[c * 4 + 0], w1 = conv_w[c * 4 + 1];
    float w2 = conv_w[c * 4 + 2], w3 = conv_w[c * 4 + 3];
    float cb = conv_b[c];
    _Float16* xcrow = xc_g + (size_t)bn * LSEQ * DFF;
    float p0 = 0.f, p1 = 0.f, p2 = 0.f; // x[l-3], x[l-2], x[l-1]
    for (int l = 0; l < LSEQ; ++l) {
      float cur = (float)sh_xin[l * DFF + c];
      float a = cb + w0 * p0 + w1 * p1 + w2 * p2 + w3 * cur;
      xcrow[l * DFF + c] = (_Float16)silu_f(a);
      p0 = p1; p1 = p2; p2 = cur;
    }
  }
}

// ---------------------------------------------------------------------------
// K2: async-stage xc -> LDS, x_proj GEMM (64x48,K=256) + dt_proj GEMM
//     (64x256,K=16 padded) + softplus + selective scan + output activation
// ---------------------------------------------------------------------------
__global__ __launch_bounds__(256) void k_ssm(
    const _Float16* __restrict__ xc_g, const _Float16* __restrict__ Wx,
    const _Float16* __restrict__ Wdt, const float* __restrict__ dt_b,
    const float* __restrict__ A_log, const float* __restrict__ Dp,
    const _Float16* __restrict__ zs_g, _Float16* __restrict__ yact_g) {
  __shared__ _Float16 s_xc[LSEQ * DFF];    // 32 KB conv output tile
  __shared__ float    s_dbc[LSEQ][48];     // 12 KB: [dr(16) | B(16) | C(16)]
  __shared__ _Float16 s_delta[LSEQ * DFF]; // 32 KB
  const int bn = blockIdx.x, tid = threadIdx.x;
  const int wv = tid >> 5, lane = tid & 31;
  const int hs = lane >> 4, nn = lane & 15;

  // async-copy the whole xc tile for this sequence into LDS (ASYNCcnt path)
  async_stage_tile(xc_g + (size_t)bn * LSEQ * DFF, s_xc, (LSEQ * DFF) / 8);
  __syncthreads();

  // x_proj: 12 tiles (4 rt x 3 ct) over 8 waves, A from LDS
  for (int t = wv; t < 12; t += 8) {
    const int rt = t / 3, ct = t % 3;
    v8f acc = {};
    for (int kk = 0; kk < 8; ++kk) {
      v16h a = load_a_frag<DFF>(s_xc, rt * 16, kk * 32);
      v16h b = load_b_frag<DFF>(Wx, ct * 16, kk * 32);
      acc = wmma_f16(a, b, acc);
    }
#pragma unroll
    for (int r = 0; r < 8; ++r)
      s_dbc[rt * 16 + r + 8 * hs][ct * 16 + nn] = acc[r];
  }
  __syncthreads();

  // dt_proj: M=64, N=256, K=16 zero-padded to 32; wave w owns ct = 2w, 2w+1
  for (int ct = wv * 2; ct < wv * 2 + 2; ++ct) {
#pragma unroll
    for (int rt = 0; rt < 4; ++rt) {
      V16H a;
      const int m = rt * 16 + nn;
#pragma unroll
      for (int v = 0; v < 8; ++v) {
        int kb = 2 * v + 16 * hs;
        a.h[2 * v]     = (kb     < DTR_) ? (_Float16)s_dbc[m][kb]     : (_Float16)0.f;
        a.h[2 * v + 1] = (kb + 1 < DTR_) ? (_Float16)s_dbc[m][kb + 1] : (_Float16)0.f;
      }
      V16H b;
      if (hs == 0) { // lanes 16-31 cover K=16..31 -> zero pad
        const uint4* p = (const uint4*)(Wdt + (size_t)(ct * 16 + nn) * DTR_);
        uint4 q0 = p[0], q1 = p[1];
        b.u[0] = q0.x; b.u[1] = q0.y; b.u[2] = q0.z; b.u[3] = q0.w;
        b.u[4] = q1.x; b.u[5] = q1.y; b.u[6] = q1.z; b.u[7] = q1.w;
      } else {
#pragma unroll
        for (int v = 0; v < 8; ++v) b.u[v] = 0u;
      }
      v8f acc = {};
      acc = wmma_f16(a.v, b.v, acc);
#pragma unroll
      for (int r = 0; r < 8; ++r) {
        int mm = rt * 16 + r + 8 * hs;
        int n  = ct * 16 + nn;
        s_delta[mm * DFF + n] = (_Float16)softplus_f(acc[r] + dt_b[n]);
      }
    }
  }
  __syncthreads();

  // selective scan: thread owns channel c, 16-state recurrence over l
  {
    const int c = tid;
    float An[DS_], h[DS_];
#pragma unroll
    for (int s = 0; s < DS_; ++s) { An[s] = -__expf(A_log[c * DS_ + s]); h[s] = 0.f; }
    const float Dc = Dp[c];
    const _Float16* zrow = zs_g + (size_t)bn * LSEQ * DFF;
    _Float16* yrow = yact_g + (size_t)bn * LSEQ * DFF;
    for (int l = 0; l < LSEQ; ++l) {
      float d  = (float)s_delta[l * DFF + c];
      float xv = (float)s_xc[l * DFF + c];
      float dx = d * xv;
      float y = 0.f;
#pragma unroll
      for (int s = 0; s < DS_; ++s) {
        float dA = __expf(d * An[s]);
        h[s] = h[s] * dA + dx * s_dbc[l][16 + s]; // B broadcast from LDS
        y += h[s] * s_dbc[l][32 + s];             // C broadcast from LDS
      }
      float zv = (float)zrow[l * DFF + c];
      yrow[l * DFF + c] = (_Float16)((y + Dc * xv) * zv);
    }
  }
}

// ---------------------------------------------------------------------------
// K3: async-stage y_act -> LDS, out_proj GEMM (64x128, K=256) + bias + residual
// ---------------------------------------------------------------------------
__global__ __launch_bounds__(256) void k_outproj(
    const _Float16* __restrict__ yact_g, const _Float16* __restrict__ Wout,
    const float* __restrict__ out_b, const float* __restrict__ x,
    float* __restrict__ out) {
  __shared__ _Float16 s_y[LSEQ * DFF]; // 32 KB
  const int bn = blockIdx.x, tid = threadIdx.x;
  const int wv = tid >> 5, lane = tid & 31;

  async_stage_tile(yact_g + (size_t)bn * LSEQ * DFF, s_y, (LSEQ * DFF) / 8);
  __syncthreads();

  const int n0 = wv * 16; // wave w owns 16 output cols
  v8f acc[4] = {};
  for (int kk = 0; kk < 8; ++kk) {
    v16h b = load_b_frag<DFF>(Wout, n0, kk * 32);
#pragma unroll
    for (int rt = 0; rt < 4; ++rt) {
      v16h a = load_a_frag<DFF>(s_y, rt * 16, kk * 32);
      acc[rt] = wmma_f16(a, b, acc[rt]);
    }
  }
  const int hs = lane >> 4, nn = lane & 15;
  const float* xrow = x + (size_t)bn * LSEQ * DM;
  float* orow = out + (size_t)bn * LSEQ * DM;
#pragma unroll
  for (int rt = 0; rt < 4; ++rt)
#pragma unroll
    for (int r = 0; r < 8; ++r) {
      int m = rt * 16 + r + 8 * hs;
      int n = n0 + nn;
      orow[m * DM + n] = acc[rt][r] + out_b[n] + xrow[m * DM + n];
    }
}

// ---------------------------------------------------------------------------
extern "C" void kernel_launch(void* const* d_in, const int* in_sizes, int n_in,
                              void* d_out, int out_size, void* d_ws, size_t ws_size,
                              hipStream_t stream) {
  (void)in_sizes; (void)n_in; (void)out_size; (void)ws_size;
  const float* x          = (const float*)d_in[0];
  const float* norm_w     = (const float*)d_in[1];
  const float* in_proj_w  = (const float*)d_in[2];
  const float* in_proj_b  = (const float*)d_in[3];
  const float* conv_w     = (const float*)d_in[4];
  const float* conv_b     = (const float*)d_in[5];
  const float* x_proj_w   = (const float*)d_in[6];
  const float* dt_proj_w  = (const float*)d_in[7];
  const float* dt_proj_b  = (const float*)d_in[8];
  const float* A_log      = (const float*)d_in[9];
  const float* Dp         = (const float*)d_in[10];
  const float* out_proj_w = (const float*)d_in[11];
  const float* out_proj_b = (const float*)d_in[12];
  float* out = (float*)d_out;

  // workspace layout (bytes)
  char* ws = (char*)d_ws;
  const size_t ACT = (size_t)BN * LSEQ * DFF * sizeof(_Float16); // 33.5 MB each
  _Float16* W16in  = (_Float16*)(ws + 0);        // 512x128
  _Float16* W16x   = (_Float16*)(ws + 131072);   // 48x256
  _Float16* W16dt  = (_Float16*)(ws + 155648);   // 256x16
  _Float16* W16out = (_Float16*)(ws + 163840);   // 128x256
  _Float16* xc16   = (_Float16*)(ws + 229376);
  _Float16* zs16   = (_Float16*)(ws + 229376 + ACT);
  _Float16* ya16   = (_Float16*)(ws + 229376 + 2 * ACT);

  k_convert<<<256, 256, 0, stream>>>(in_proj_w, 2 * DFF * DM, W16in);
  k_convert<<<48,  256, 0, stream>>>(x_proj_w, (DTR_ + 2 * DS_) * DFF, W16x);
  k_convert<<<16,  256, 0, stream>>>(dt_proj_w, DFF * DTR_, W16dt);
  k_convert<<<128, 256, 0, stream>>>(out_proj_w, DM * DFF, W16out);

  k_fused_inproj<<<BN, 256, 0, stream>>>(x, norm_w, W16in, in_proj_b,
                                         conv_w, conv_b, xc16, zs16);
  k_ssm<<<BN, 256, 0, stream>>>(xc16, W16x, W16dt, dt_proj_b, A_log, Dp,
                                zs16, ya16);
  k_outproj<<<BN, 256, 0, stream>>>(ya16, W16out, out_proj_b, x, out);
}
